// QMLClassifier_3401614098540
// MI455X (gfx1250) — matrix operand order repaired and checked by reading, not compiled, not run on hardware
//
#include <hip/hip_runtime.h>

typedef _Float16 half_t;
typedef __attribute__((ext_vector_type(16))) _Float16 v16h;
typedef __attribute__((ext_vector_type(8)))  _Float16 v8h;
typedef __attribute__((ext_vector_type(8)))  float    v8f;
typedef __attribute__((ext_vector_type(4)))  float    v4f;

#define NQ   8
#define DIM  256
#define MT   32        // batch rows per block
#define AS   264       // A-tile LDS stride in halfs (padded: bank-conflict-free b128 reads)
#define SS   257       // state-tile LDS stride in floats

// ---------------------------------------------------------------------------
// Kernel 1: build U^T (re/im) in f16. One block per basis state j; 256 threads
// simulate the 46-gate circuit on that basis vector in LDS, then scatter the
// final amplitudes into Bt[n=amp][k=basis] layout (what the GEMM B-operand wants).
// ---------------------------------------------------------------------------
__global__ __launch_bounds__(DIM)
void build_unitary(const float* __restrict__ qw,
                   half_t* __restrict__ bt_re,
                   half_t* __restrict__ bt_im) {
  __shared__ float sr[DIM];
  __shared__ float si[DIM];
  const int t = threadIdx.x;   // amplitude index
  const int j = blockIdx.x;    // basis state
  sr[t] = (t == j) ? 1.0f : 0.0f;
  si[t] = 0.0f;
  __syncthreads();

  for (int l = 0; l < 2; ++l) {
    for (int i = 0; i < NQ; ++i) {
      const int mask = 1 << (7 - i);
      const float ty = qw[(l * 8 + i) * 2 + 0];
      const float tz = qw[(l * 8 + i) * 2 + 1];
      // RY(ty): [[c,-s],[s,c]] on bit (7-i)
      {
        const float c = cosf(ty * 0.5f), s = sinf(ty * 0.5f);
        if ((t & mask) == 0) {
          const int k1 = t | mask;
          const float ar = sr[t],  ai = si[t];
          const float br = sr[k1], bi = si[k1];
          sr[t]  = c * ar - s * br;  si[t]  = c * ai - s * bi;
          sr[k1] = s * ar + c * br;  si[k1] = s * ai + c * bi;
        }
      }
      __syncthreads();
      // RZ(tz): diag(e, conj(e)), e = cos(tz/2) - i*sin(tz/2)
      {
        const float cz = cosf(tz * 0.5f), sz = sinf(tz * 0.5f);
        const float er = cz;
        const float ei = (t & mask) ? sz : -sz;
        const float ar = sr[t], ai = si[t];
        sr[t] = ar * er - ai * ei;
        si[t] = ar * ei + ai * er;
      }
      __syncthreads();
    }
    for (int i = 0; i < NQ - 1; ++i) {   // CNOT chain (control i, target i+1)
      const int cm = 1 << (7 - i), tm = 1 << (6 - i);
      if ((t & cm) && !(t & tm)) {
        const int k1 = t | tm;
        const float ar = sr[t], ai = si[t];
        sr[t] = sr[k1];  si[t] = si[k1];
        sr[k1] = ar;     si[k1] = ai;
      }
      __syncthreads();
    }
  }
  // Bt[n][k] = U[n][k]  (n = final amplitude index, k = input basis index)
  bt_re[t * DIM + j] = (half_t)sr[t];
  bt_im[t * DIM + j] = (half_t)si[t];
}

// ---------------------------------------------------------------------------
// Kernel 2: fused  S = X @ U^T  (WMMA f16->f32)  + measurements + BN + MLP.
// Block: 256 threads (8 waves), 32 batch rows. N = 256(re)+256(im) = 32 tiles.
// ---------------------------------------------------------------------------
__global__ __launch_bounds__(256)
void qml_main(const float* __restrict__ X,
              const half_t* __restrict__ bt_re,
              const half_t* __restrict__ bt_im,
              const float* __restrict__ g,  const float* __restrict__ bbn,
              const float* __restrict__ rm, const float* __restrict__ rv,
              const float* __restrict__ W1, const float* __restrict__ b1,
              const float* __restrict__ W2, const float* __restrict__ b2,
              float* __restrict__ out) {
  __shared__ __align__(16) half_t aT[MT * AS];     // f16 A tile  [32][256] padded
  __shared__ float Sre[MT * SS];                   // state re    [32][256] padded
  __shared__ float Sim[MT * SS];                   // state im
  __shared__ float prt[MT][8];                     // ||x||^2 partials
  __shared__ float red[MT][8][16];                 // measurement partials

  const int tid  = threadIdx.x;
  const int row0 = blockIdx.x * MT;

  // ---- Stage A tile: fp32 -> f16 into LDS, accumulate fp32 norm^2 partials
  {
    const int r   = tid >> 3;        // 0..31
    const int cch = tid & 7;         // 8 chunks of 32 cols
    const float* xp = X + (size_t)(row0 + r) * DIM + cch * 32;
    half_t* ap = aT + r * AS + cch * 32;
    float acc = 0.0f;
#pragma unroll
    for (int q = 0; q < 8; ++q) {
      const v4f v = *(const v4f*)(xp + q * 4);
      acc += v[0]*v[0] + v[1]*v[1] + v[2]*v[2] + v[3]*v[3];
      ap[q*4+0] = (half_t)v[0]; ap[q*4+1] = (half_t)v[1];
      ap[q*4+2] = (half_t)v[2]; ap[q*4+3] = (half_t)v[3];
    }
    prt[r][cch] = acc;
  }
  __syncthreads();

  // ---- GEMM: wave w owns M-tile (w&1) and 8 N-tiles (j = (w>>1)*8 .. +7)
  {
    const int wave = tid >> 5;
    const int lane = tid & 31;
    const int m    = wave & 1;
    const int jset = wave >> 1;
    const int lr   = lane & 15;      // N / M index within tile
    const int lh   = lane >> 4;      // K-half selector

    for (int jj = 0; jj < 8; ++jj) {
      const int j = jset * 8 + jj;                 // 0..15 re, 16..31 im
      const half_t* Bmat = (j < 16) ? bt_re : bt_im;
      const int nt = (j & 15) * 16;
      v8f acc = {0.f, 0.f, 0.f, 0.f, 0.f, 0.f, 0.f, 0.f};
#pragma unroll
      for (int kk = 0; kk < 8; ++kk) {
        const int kb = kk * 32;
        // A fragment: lane&15 = M row; halves K = kb+lh*8..+7 and kb+16+lh*8..+7
        const half_t* pa = aT + (m * 16 + lr) * AS + kb + lh * 8;
        const v8h a0 = *(const v8h*)pa;
        const v8h a1 = *(const v8h*)(pa + 16);
        v16h a;
#pragma unroll
        for (int h = 0; h < 8; ++h) { a[h] = a0[h]; a[h + 8] = a1[h]; }
        // B fragment: lane&15 = N col; halves K = kb+lh*8..+7 and kb+16+lh*8..+7 (Bt[n][k])
        const half_t* pb = Bmat + (size_t)(nt + lr) * DIM + kb + lh * 8;
        const v8h f0 = *(const v8h*)pb;
        const v8h f1 = *(const v8h*)(pb + 16);
        v16h bf;
#pragma unroll
        for (int h = 0; h < 8; ++h) { bf[h] = f0[h]; bf[h + 8] = f1[h]; }
        acc = __builtin_amdgcn_wmma_f32_16x16x32_f16(
                  false, a, false, bf, (short)0, acc, false, false);
      }
      // D layout: vgpr r -> row r + 8*lh, col = nt + (lane&15)
      float* Sm = (j < 16) ? Sre : Sim;
      const int col = nt + lr;
#pragma unroll
      for (int r8 = 0; r8 < 8; ++r8)
        Sm[(m * 16 + r8 + lh * 8) * SS + col] = acc[r8];
    }
  }
  __syncthreads();

  // ---- Measurements: 8 threads per row, partial Z and X expectations
  {
    const int r = tid >> 3;
    const int t = tid & 7;
    const float* sre = Sre + r * SS;
    const float* sim = Sim + r * SS;
    float z[8] = {0, 0, 0, 0, 0, 0, 0, 0};
    for (int kk = 0; kk < 32; ++kk) {
      const int k = t * 32 + kk;
      const float p = sre[k] * sre[k] + sim[k] * sim[k];
#pragma unroll
      for (int i = 0; i < 8; ++i)
        z[i] += ((k >> (7 - i)) & 1) ? -p : p;
    }
    float xv[8];
#pragma unroll
    for (int i = 0; i < 8; ++i) {
      const int mask = 1 << (7 - i);
      float acc = 0.0f;
      for (int idx = t * 16; idx < t * 16 + 16; ++idx) {
        const int k0 = ((idx & ~(mask - 1)) << 1) | (idx & (mask - 1));
        const int k1 = k0 | mask;
        acc += sre[k0] * sre[k1] + sim[k0] * sim[k1];
      }
      xv[i] = acc;
    }
#pragma unroll
    for (int i = 0; i < 8; ++i) { red[r][t][i] = z[i]; red[r][t][8 + i] = xv[i]; }
  }
  __syncthreads();

  // ---- Per-row tail: reduce, scale by 1/||x||^2, BN, MLP 16->8->2
  if (tid < MT) {
    const int r = tid;
    float n2 = 0.0f;
    for (int t = 0; t < 8; ++t) n2 += prt[r][t];
    const float inv = 1.0f / n2;
    float q[16];
    for (int i = 0; i < 16; ++i) {
      float s = 0.0f;
      for (int t = 0; t < 8; ++t) s += red[r][t][i];
      q[i] = (i < 8 ? s : 2.0f * s) * inv;
    }
    float h[16];
    for (int i = 0; i < 16; ++i)
      h[i] = g[i] * (q[i] - rm[i]) * rsqrtf(rv[i] + 1e-5f) + bbn[i];
    float a1[8];
    for (int jj = 0; jj < 8; ++jj) {
      float s = b1[jj];
      for (int i = 0; i < 16; ++i) s += h[i] * W1[i * 8 + jj];
      a1[jj] = s > 0.0f ? s : 0.0f;
    }
    for (int o = 0; o < 2; ++o) {
      float s = b2[o];
      for (int jj = 0; jj < 8; ++jj) s += a1[jj] * W2[jj * 2 + o];
      out[(size_t)(row0 + r) * 2 + o] = s;
    }
  }
}

extern "C" void kernel_launch(void* const* d_in, const int* in_sizes, int n_in,
                              void* d_out, int out_size, void* d_ws, size_t ws_size,
                              hipStream_t stream) {
  const float* x  = (const float*)d_in[0];
  const float* qw = (const float*)d_in[1];
  const float* g  = (const float*)d_in[2];
  const float* b  = (const float*)d_in[3];
  const float* rm = (const float*)d_in[4];
  const float* rv = (const float*)d_in[5];
  const float* W1 = (const float*)d_in[6];
  const float* b1 = (const float*)d_in[7];
  const float* W2 = (const float*)d_in[8];
  const float* b2 = (const float*)d_in[9];
  float* out = (float*)d_out;

  half_t* bt_re = (half_t*)d_ws;              // 256*256 f16 = 128 KB
  half_t* bt_im = bt_re + DIM * DIM;          // +128 KB

  const int B = in_sizes[0] / DIM;            // 65536

  build_unitary<<<DIM, DIM, 0, stream>>>(qw, bt_re, bt_im);
  qml_main<<<B / MT, 256, 0, stream>>>(x, bt_re, bt_im,
                                       g, b, rm, rv, W1, b1, W2, b2, out);
}
